// MultiheadSelfAttention_26723286516123
// MI455X (gfx1250) — matrix operand order repaired and checked by reading, not compiled, not run on hardware
//
#include <hip/hip_runtime.h>

typedef __attribute__((ext_vector_type(16))) __bf16 v16bf;
typedef __attribute__((ext_vector_type(8)))  __bf16 v8bf;
typedef __attribute__((ext_vector_type(4)))  __bf16 v4bf;
typedef __attribute__((ext_vector_type(8)))  float  v8f;
typedef __attribute__((ext_vector_type(4)))  float  v4f;

union Frag16 { v16bf v; v8bf h[2]; };
union FragC  { v8f   v; float f[8]; };

constexpr int Bb = 4, Ss = 2048, Dd = 1024, Hh = 16, HD = 64;
constexpr int M_TOK = Bb * Ss;           // 8192 tokens

// ---------------------------------------------------------------------------
// fp32 -> bf16 conversion (vectorized: float4 in, 4x bf16 out)
// ---------------------------------------------------------------------------
__global__ void cvt_f32_to_bf16(const float* __restrict__ in,
                                __bf16* __restrict__ out, int n4) {
  int i = blockIdx.x * blockDim.x + threadIdx.x;
  int stride = gridDim.x * blockDim.x;
  for (; i < n4; i += stride) {
    v4f f = *(const v4f*)(in + 4 * (size_t)i);
    v4bf o;
    o[0] = (__bf16)f[0]; o[1] = (__bf16)f[1];
    o[2] = (__bf16)f[2]; o[3] = (__bf16)f[3];
    *(v4bf*)(out + 4 * (size_t)i) = o;
  }
}

// ---------------------------------------------------------------------------
// WMMA fragment helpers (layouts per CDNA5 ISA 7.12.2, wave32)
// ---------------------------------------------------------------------------
static __device__ __forceinline__
v16bf load_a(const __bf16* __restrict__ base, int ld, int m0, int k0, int lane) {
  int r    = m0 + (lane & 15);
  int koff = (lane >> 4) * 8;
  const __bf16* p = base + (size_t)r * ld + k0 + koff;
  Frag16 f;
  f.h[0] = *(const v8bf*)(p);
  f.h[1] = *(const v8bf*)(p + 16);
  return f.v;
}

static __device__ __forceinline__
v16bf load_b(const __bf16* __restrict__ base, int ld, int n0, int k0, int lane) {
  int n    = n0 + (lane & 15);
  int koff = (lane >> 4) * 8;
  const __bf16* p = base + (size_t)n * ld + k0 + koff;
  Frag16 f;
  f.h[0] = *(const v8bf*)(p);
  f.h[1] = *(const v8bf*)(p + 16);
  return f.v;
}

// B fragment sourced from an LDS tile (rows padded to LDS_STRIDE elems)
constexpr int LDS_STRIDE = 40;                       // 80B rows: conflict-free
static __device__ __forceinline__
v16bf load_b_lds(const __bf16* base, int n0, int lane) {
  const __bf16* p = base + (size_t)(n0 + (lane & 15)) * LDS_STRIDE + (lane >> 4) * 8;
  Frag16 f;
  f.h[0] = *(const v8bf*)(p);
  f.h[1] = *(const v8bf*)(p + 16);
  return f.v;
}

static __device__ __forceinline__
v8f wmma_bf16(v16bf a, v16bf b, v8f c) {
  return __builtin_amdgcn_wmma_f32_16x16x32_bf16(
      false, a, false, b, (short)0, c, false, false);
}

// Async copy of one 16B chunk global -> LDS (gfx1250, tracked by ASYNCcnt).
// ldsOff is derived via ptrtoint from the real LDS object (escape => the
// "memory" clobber is known to cover the LDS buffer).
static __device__ __forceinline__
void async_g2l_b128(unsigned ldsOff, const void* gsrc) {
  asm volatile("global_load_async_to_lds_b128 %0, %1, off"
               :: "v"(ldsOff), "v"((unsigned long long)(uintptr_t)gsrc)
               : "memory");
}
static __device__ __forceinline__ void wait_async0() {
  asm volatile("s_wait_asynccnt 0x0" ::: "memory");
}

// ---------------------------------------------------------------------------
// Shared GEMM inner pipeline: block = 8 waves, block tile 256(M) x 64(N).
// Wave tile 32(M) x 64(N): 2 A-fragments x 4 LDS B-fragments = 8 WMMA per
// 32-k half-step.  W tile double-buffered in LDS via async-to-LDS; A
// fragments software-pipelined from global.  B fragments loaded into
// separate regs before the WMMA group so ds_loads clause with batched waits.
// ---------------------------------------------------------------------------
struct GemmAcc { v8f a[2][4]; };

static __device__ __forceinline__
GemmAcc gemm_core(const __bf16* __restrict__ Amat, const __bf16* __restrict__ W,
                  __bf16 (*ldsB)[64 * LDS_STRIDE], int mW, int n0,
                  int tid, int lane) {
  constexpr unsigned BUF_BYTES = 64 * LDS_STRIDE * 2;          // 5120
  const int srow = tid >> 2, schk = tid & 3;                   // 64x32 W tile
  const __bf16* gsrc0 = W + (size_t)(n0 + srow) * Dd + schk * 8;
  const unsigned dstA =
      (unsigned)(uintptr_t)(void*)&ldsB[0][srow * LDS_STRIDE + schk * 8];
  const unsigned dstB = dstA + BUF_BYTES;
  const __bf16* bb0 = &ldsB[0][0];
  const __bf16* bb1 = &ldsB[1][0];

  GemmAcc g;
  v8f zero = {};
#pragma unroll
  for (int mi = 0; mi < 2; ++mi)
#pragma unroll
    for (int j = 0; j < 4; ++j) g.a[mi][j] = zero;

  constexpr int NK = Dd / 32;                                  // 32 (even)
  async_g2l_b128(dstA, gsrc0);
  v16bf a0[2], a1[2];
#pragma unroll
  for (int mi = 0; mi < 2; ++mi) {
    a0[mi] = load_a(Amat, Dd, mW + mi * 16, 0, lane);
    a1[mi] = a0[mi];
  }

  for (int i = 0; i < NK; i += 2) {
    // ---- half 0: consume buf0/a0, prefetch buf1/a1 ----
    wait_async0();
    __syncthreads();
    if (i + 1 < NK) {
      async_g2l_b128(dstB, gsrc0 + (i + 1) * 32);
#pragma unroll
      for (int mi = 0; mi < 2; ++mi)
        a1[mi] = load_a(Amat, Dd, mW + mi * 16, (i + 1) * 32, lane);
    }
    {
      v16bf b0 = load_b_lds(bb0,  0, lane);
      v16bf b1 = load_b_lds(bb0, 16, lane);
      v16bf b2 = load_b_lds(bb0, 32, lane);
      v16bf b3 = load_b_lds(bb0, 48, lane);
#pragma unroll
      for (int mi = 0; mi < 2; ++mi) {
        g.a[mi][0] = wmma_bf16(a0[mi], b0, g.a[mi][0]);
        g.a[mi][1] = wmma_bf16(a0[mi], b1, g.a[mi][1]);
        g.a[mi][2] = wmma_bf16(a0[mi], b2, g.a[mi][2]);
        g.a[mi][3] = wmma_bf16(a0[mi], b3, g.a[mi][3]);
      }
    }
    if (i + 1 >= NK) break;

    // ---- half 1: consume buf1/a1, prefetch buf0/a0 ----
    wait_async0();
    __syncthreads();
    if (i + 2 < NK) {
      async_g2l_b128(dstA, gsrc0 + (i + 2) * 32);
#pragma unroll
      for (int mi = 0; mi < 2; ++mi)
        a0[mi] = load_a(Amat, Dd, mW + mi * 16, (i + 2) * 32, lane);
    }
    {
      v16bf b0 = load_b_lds(bb1,  0, lane);
      v16bf b1 = load_b_lds(bb1, 16, lane);
      v16bf b2 = load_b_lds(bb1, 32, lane);
      v16bf b3 = load_b_lds(bb1, 48, lane);
#pragma unroll
      for (int mi = 0; mi < 2; ++mi) {
        g.a[mi][0] = wmma_bf16(a1[mi], b0, g.a[mi][0]);
        g.a[mi][1] = wmma_bf16(a1[mi], b1, g.a[mi][1]);
        g.a[mi][2] = wmma_bf16(a1[mi], b2, g.a[mi][2]);
        g.a[mi][3] = wmma_bf16(a1[mi], b3, g.a[mi][3]);
      }
    }
  }
  return g;
}

// ---------------------------------------------------------------------------
// GEMM 1: qkv = x @ W_qkv^T  (M=8192, N=3072, K=1024)
// Epilogue scatters to Q[b,h,s,d], K[b,h,s,d], V^T[b,h,d,s] (bf16).
// ---------------------------------------------------------------------------
__global__ void __launch_bounds__(256)
gemm_qkv(const __bf16* __restrict__ X, const __bf16* __restrict__ W,
         __bf16* __restrict__ Qb, __bf16* __restrict__ Kb,
         __bf16* __restrict__ Vt) {
  __shared__ __align__(16) __bf16 ldsB[2][64 * LDS_STRIDE];

  const int tid  = threadIdx.x;
  const int lane = tid & 31;
  const int wave = tid >> 5;
  const int NT   = 3072 / 64;                                  // 48
  const int mB   = (blockIdx.x / NT) * 256;
  const int n0   = (blockIdx.x % NT) * 64;
  const int mW   = mB + wave * 32;

  GemmAcc g = gemm_core(X, W, ldsB, mW, n0, tid, lane);

  const int lhalf = lane >> 4, lcol = lane & 15;
#pragma unroll
  for (int mi = 0; mi < 2; ++mi) {
#pragma unroll
    for (int j = 0; j < 4; ++j) {
      FragC c; c.v = g.a[mi][j];
#pragma unroll
      for (int r = 0; r < 8; ++r) {
        int m = mW + mi * 16 + r + 8 * lhalf;   // token
        int n = n0 + j * 16 + lcol;             // 0..3071
        int b = m >> 11, s = m & 2047;
        int seg = n >> 10, nn = n & 1023;
        int hh = nn >> 6, dd = nn & 63;
        __bf16 val = (__bf16)c.f[r];
        size_t headTok = (size_t)(b * Hh + hh) * Ss + s;
        if (seg == 0)      Qb[headTok * HD + dd] = val;
        else if (seg == 1) Kb[headTok * HD + dd] = val;
        else               Vt[((size_t)(b * Hh + hh) * HD + dd) * Ss + s] = val;
      }
    }
  }
}

// ---------------------------------------------------------------------------
// Flash attention (causal).  One wave per (b, h, 16-query tile).
// ---------------------------------------------------------------------------
__global__ void __launch_bounds__(256)
attn_kernel(const __bf16* __restrict__ Qb, const __bf16* __restrict__ Kb,
            const __bf16* __restrict__ Vt, __bf16* __restrict__ Ob) {
  __shared__ __align__(16) __bf16 ldsP[8][16][40];

  const int lane = threadIdx.x & 31;
  const int wave = threadIdx.x >> 5;
  const int wid  = blockIdx.x * 8 + wave;   // 0..8191
  const int qt   = wid & 127;
  const int bh   = wid >> 7;                // b*16 + h
  const int lhalf = lane >> 4, lcol = lane & 15;

  const __bf16* Qh = Qb + (size_t)bh * Ss * HD;
  const __bf16* Kh = Kb + (size_t)bh * Ss * HD;
  const __bf16* Vh = Vt + (size_t)bh * HD * Ss;

  v16bf qf[2];
#pragma unroll
  for (int c = 0; c < 2; ++c) qf[c] = load_a(Qh, HD, qt * 16, c * 32, lane);

  v8f zero = {};
  FragC o[4];
#pragma unroll
  for (int j = 0; j < 4; ++j) o[j].v = zero;
  float mrow[8], lrow[8];
#pragma unroll
  for (int r = 0; r < 8; ++r) { mrow[r] = -1e30f; lrow[r] = 0.f; }

  const int q_lo = qt * 16;
  const int nkb  = (q_lo + 15) / 32 + 1;

  for (int kb = 0; kb < nkb; ++kb) {
    // ---- scores: Q (16x64) x K^T (64x32) ----
    FragC s[2];
#pragma unroll
    for (int t = 0; t < 2; ++t) {
      v8f acc = zero;
#pragma unroll
      for (int c = 0; c < 2; ++c)
        acc = wmma_bf16(qf[c], load_b(Kh, HD, kb * 32 + t * 16, c * 32, lane), acc);
      s[t].v = acc;
    }

    // ---- V fragments early: latency hidden under softmax VALU work ----
    v16bf vf[4];
#pragma unroll
    for (int j = 0; j < 4; ++j) vf[j] = load_b(Vh, Ss, j * 16, kb * 32, lane);

    // ---- scale + causal mask ----
    const bool needMask = (kb * 32 + 31) > q_lo;
#pragma unroll
    for (int t = 0; t < 2; ++t) {
      int key = kb * 32 + t * 16 + lcol;
#pragma unroll
      for (int r = 0; r < 8; ++r) {
        float v = s[t].f[r] * 0.125f;
        int row = q_lo + r + 8 * lhalf;
        if (needMask && key > row) v = -1e30f;
        s[t].f[r] = v;
      }
    }

    // ---- online softmax (row reductions across 16-lane halves) ----
#pragma unroll
    for (int r = 0; r < 8; ++r) {
      float tmax = fmaxf(s[0].f[r], s[1].f[r]);
#pragma unroll
      for (int off = 1; off < 16; off <<= 1)
        tmax = fmaxf(tmax, __shfl_xor(tmax, off, 32));
      float mn    = fmaxf(mrow[r], tmax);
      float alpha = __expf(mrow[r] - mn);
      float p0 = __expf(s[0].f[r] - mn);
      float p1 = __expf(s[1].f[r] - mn);
      s[0].f[r] = p0; s[1].f[r] = p1;
      float tsum = p0 + p1;
#pragma unroll
      for (int off = 1; off < 16; off <<= 1)
        tsum += __shfl_xor(tsum, off, 32);
      lrow[r] = lrow[r] * alpha + tsum;
      mrow[r] = mn;
#pragma unroll
      for (int j = 0; j < 4; ++j) o[j].f[r] *= alpha;
    }

    // ---- repack P (C layout -> A layout) via per-wave LDS tile ----
#pragma unroll
    for (int t = 0; t < 2; ++t)
#pragma unroll
      for (int r = 0; r < 8; ++r)
        ldsP[wave][r + 8 * lhalf][t * 16 + lcol] = (__bf16)s[t].f[r];
    asm volatile("s_wait_dscnt 0" ::: "memory");   // same-wave DS RAW
    Frag16 pf;
    {
      const __bf16* p = &ldsP[wave][lcol][lhalf * 8];
      pf.h[0] = *(const v8bf*)(p);
      pf.h[1] = *(const v8bf*)(p + 16);
    }

    // ---- O += P (16x32) x V (32x64) ----
#pragma unroll
    for (int j = 0; j < 4; ++j)
      o[j].v = wmma_bf16(pf.v, vf[j], o[j].v);
  }

  // ---- epilogue: normalize, store attn[b, s, h*64+d] (bf16) ----
  const int b = bh >> 4, h = bh & 15;
#pragma unroll
  for (int j = 0; j < 4; ++j) {
#pragma unroll
    for (int r = 0; r < 8; ++r) {
      int srow = q_lo + r + 8 * lhalf;
      float val = o[j].f[r] / lrow[r];
      Ob[(size_t)(b * Ss + srow) * Dd + h * HD + j * 16 + lcol] = (__bf16)val;
    }
  }
}

// ---------------------------------------------------------------------------
// GEMM 2: out = attn @ W_out^T  (M=8192, N=1024, K=1024, f32 out)
// ---------------------------------------------------------------------------
__global__ void __launch_bounds__(256)
gemm_out(const __bf16* __restrict__ A, const __bf16* __restrict__ W,
         float* __restrict__ out) {
  __shared__ __align__(16) __bf16 ldsB[2][64 * LDS_STRIDE];

  const int tid  = threadIdx.x;
  const int lane = tid & 31;
  const int wave = tid >> 5;
  const int NT   = 1024 / 64;                                  // 16
  const int mB   = (blockIdx.x / NT) * 256;
  const int n0   = (blockIdx.x % NT) * 64;
  const int mW   = mB + wave * 32;

  GemmAcc g = gemm_core(A, W, ldsB, mW, n0, tid, lane);

  const int lhalf = lane >> 4, lcol = lane & 15;
#pragma unroll
  for (int mi = 0; mi < 2; ++mi) {
#pragma unroll
    for (int j = 0; j < 4; ++j) {
      FragC c; c.v = g.a[mi][j];
#pragma unroll
      for (int r = 0; r < 8; ++r) {
        int m = mW + mi * 16 + r + 8 * lhalf;
        int n = n0 + j * 16 + lcol;
        out[(size_t)m * Dd + n] = c.f[r];
      }
    }
  }
}

// ---------------------------------------------------------------------------
// Host launcher
// ---------------------------------------------------------------------------
extern "C" void kernel_launch(void* const* d_in, const int* in_sizes, int n_in,
                              void* d_out, int out_size, void* d_ws, size_t ws_size,
                              hipStream_t stream) {
  const float* x    = (const float*)d_in[0];   // [4, 2048, 1024]
  const float* Wqkv = (const float*)d_in[1];   // [3072, 1024]
  const float* Wout = (const float*)d_in[2];   // [1024, 1024]
  float* out = (float*)d_out;                  // [4, 2048, 1024]

  __bf16* p = (__bf16*)d_ws;
  __bf16* xb    = p; p += (size_t)M_TOK * Dd;
  __bf16* wqkvb = p; p += (size_t)3 * Dd * Dd;
  __bf16* woutb = p; p += (size_t)Dd * Dd;
  __bf16* Qb    = p; p += (size_t)M_TOK * Dd;
  __bf16* Kb    = p; p += (size_t)M_TOK * Dd;
  __bf16* Vt    = p; p += (size_t)M_TOK * Dd;
  __bf16* Ab    = p; p += (size_t)M_TOK * Dd;   // ~88 MiB total

  cvt_f32_to_bf16<<<512, 256, 0, stream>>>(x,    xb,    M_TOK * Dd / 4);
  cvt_f32_to_bf16<<<512, 256, 0, stream>>>(Wqkv, wqkvb, 3 * Dd * Dd / 4);
  cvt_f32_to_bf16<<<512, 256, 0, stream>>>(Wout, woutb, Dd * Dd / 4);

  gemm_qkv   <<<1536, 256, 0, stream>>>(xb, wqkvb, Qb, Kb, Vt);  // 256x64 blocks
  attn_kernel<<<1024, 256, 0, stream>>>(Qb, Kb, Vt, Ab);
  gemm_out   <<< 512, 256, 0, stream>>>(Ab, woutb, out);         // 256x64 blocks
}
